// SingleValueAdapterAttnProcessor_2826088480816
// MI455X (gfx1250) — matrix operand order repaired and checked by reading, not compiled, not run on hardware
//
#include <hip/hip_runtime.h>
#include <hip/hip_bf16.h>

typedef __bf16 bf16_t;
typedef bf16_t v16bf __attribute__((ext_vector_type(16)));
typedef bf16_t v8bf  __attribute__((ext_vector_type(8)));
typedef float  v8f   __attribute__((ext_vector_type(8)));

// ---------------------------------------------------------------------------
// WMMA helpers (CDNA5 gfx1250, wave32)
// ---------------------------------------------------------------------------
__device__ __forceinline__ v16bf load_a_frag(const bf16_t* p) {
  // A-matrix 16x32 bf16 fragment: per lane, two contiguous 8-element chunks
  // at k-offsets {0..7} and {16..23} relative to (kk + hi*8).
  v8bf lo = *(const v8bf*)p;
  v8bf hh = *(const v8bf*)(p + 16);
  return __builtin_shufflevector(lo, hh, 0,1,2,3,4,5,6,7,8,9,10,11,12,13,14,15);
}

__device__ __forceinline__ v8f wmma_bf16(v16bf a, v16bf b, v8f c) {
  return __builtin_amdgcn_wmma_f32_16x16x32_bf16(
      /*neg_a=*/false, a, /*neg_b=*/false, b,
      /*c_mod=*/(short)0, c, /*reuse_a=*/false, /*reuse_b=*/false);
}

// ---------------------------------------------------------------------------
// bf16 WMMA GEMM with async B staging:
//   C[M,N] = A[M,K] * Bt[N,K]^T      (Bt row-major N x K)
// Block = 256 threads = 8 waves, 128x128 tile; wave = 32x64 sub-tile.
// B panel (128 x 64 k-slice) is staged into LDS once per block via
// GLOBAL_LOAD_ASYNC_TO_LDS_B128 (ASYNCcnt), then consumed via ds_load.
// M % 128 == 0, N % 128 == 0, K % 64 == 0.
// ---------------------------------------------------------------------------
template <bool OUT_BF16>
__global__ __launch_bounds__(256)
void gemm_bf16_wmma(const bf16_t* __restrict__ A, const bf16_t* __restrict__ Bt,
                    void* __restrict__ Cout, const float* __restrict__ bias,
                    int M, int N, int K) {
  constexpr int BK  = 64;   // K elems staged per step
  constexpr int LDB = 88;   // padded LDS row stride (176B = 44 dwords, spreads banks)
  __shared__ __align__(16) bf16_t Bs[128 * LDB];

  const int tid  = threadIdx.x;
  const int lane = tid & 31;
  const int w    = tid >> 5;
  const int nl   = lane & 15;
  const int hi   = lane >> 4;
  const long m0  = (long)blockIdx.y * 128 + (w & 3) * 32;
  const int  nb  = blockIdx.x * 128;        // block N base
  const int  wn  = (w >> 2) * 64;           // wave N offset inside tile

  v8f acc[2][4] = {};
  const bf16_t* arow0 = A + (m0 + nl)      * (long)K + hi * 8;
  const bf16_t* arow1 = A + (m0 + 16 + nl) * (long)K + hi * 8;

  // async staging map: 128 rows x 8 chunks of 16B (= 64 elems); 256 thr x 4 chunks
  const int c0 = tid * 4;

  for (int kk = 0; kk < K; kk += BK) {
    // ---- async copy B panel [nb..nb+127][kk..kk+63] -> LDS ----
#pragma unroll
    for (int c = 0; c < 4; ++c) {
      int ch  = c0 + c;
      int row = ch >> 3;            // 0..127
      int ko  = (ch & 7) * 8;       // 0..56 elems
      const bf16_t* gp = Bt + (long)(nb + row) * K + kk + ko;
      unsigned ldsa = (unsigned)(size_t)(&Bs[row * LDB + ko]);
      asm volatile("global_load_async_to_lds_b128 %0, %1, off"
                   :: "v"(ldsa), "v"((unsigned long long)(size_t)gp)
                   : "memory");
    }
    // prefetch upcoming A rows while the async copy is in flight
    __builtin_prefetch(arow0 + kk + BK, 0, 0);
    __builtin_prefetch(arow1 + kk + BK, 0, 0);
    asm volatile("s_wait_asynccnt 0x0" ::: "memory");
    __syncthreads();

#pragma unroll
    for (int ks = 0; ks < BK; ks += 32) {
      v16bf a0 = load_a_frag(arow0 + kk + ks);
      v16bf a1 = load_a_frag(arow1 + kk + ks);
#pragma unroll
      for (int j = 0; j < 4; ++j) {
        v16bf b = load_a_frag(&Bs[(wn + j * 16 + nl) * LDB + ks + hi * 8]);
        acc[0][j] = wmma_bf16(a0, b, acc[0][j]);
        acc[1][j] = wmma_bf16(a1, b, acc[1][j]);
      }
    }
    __syncthreads();
  }

#pragma unroll
  for (int i = 0; i < 2; ++i)
#pragma unroll
    for (int j = 0; j < 4; ++j)
#pragma unroll
      for (int r = 0; r < 8; ++r) {
        long m = m0 + i * 16 + r + hi * 8;
        long n = (long)nb + wn + j * 16 + nl;
        float v = acc[i][j][r];
        if (OUT_BF16) {
          ((bf16_t*)Cout)[m * N + n] = (bf16_t)v;
        } else {
          ((float*)Cout)[m * N + n] = v + (bias ? bias[n] : 0.f);
        }
      }
}

// ---------------------------------------------------------------------------
// Fused cross-attention per (b, h):
//   scores[16q x 80k] = Q*K^T (WMMA), masked softmax over 77 keys,
//   P staged via LDS into A-fragment layout, out = P*V (WMMA) + v_ip.
// Grid: (S/128, H, B), 256 threads = 8 waves (16 queries each).
// ---------------------------------------------------------------------------
__global__ __launch_bounds__(256)
void attn_wmma(const bf16_t* __restrict__ Q,   // [B,4096,1280]
               const bf16_t* __restrict__ Kb,  // [B,80,1280] (rows 77..79 zero)
               const bf16_t* __restrict__ Vt,  // [B,20,64,96] (k 77..95 zero)
               const float*  __restrict__ vip, // [B,1280]
               bf16_t* __restrict__ merged) {  // [B,4096,1280]
  __shared__ __align__(16) bf16_t P[8][16][96];
  const int tid  = threadIdx.x;
  const int lane = tid & 31;
  const int w    = tid >> 5;
  const int nl   = lane & 15;
  const int hi   = lane >> 4;
  const int b    = blockIdx.z;
  const int h    = blockIdx.y;
  const int s0   = blockIdx.x * 128 + w * 16;

  // zero probability staging (covers padded keys 80..95)
  bf16_t* pf = &P[0][0][0];
  for (int i = tid; i < 8 * 16 * 96; i += 256) pf[i] = (bf16_t)0.f;
  __syncthreads();

  // ---- scores = Q K^T ----
  const bf16_t* qrow = Q + ((long)(b * 4096 + s0 + nl)) * 1280 + h * 64 + hi * 8;
  v8f sc[5] = {};
#pragma unroll
  for (int kk = 0; kk < 64; kk += 32) {
    v16bf a = load_a_frag(qrow + kk);
#pragma unroll
    for (int t = 0; t < 5; ++t) {
      v16bf bb = load_a_frag(Kb + ((long)(b * 80 + t * 16 + nl)) * 1280 +
                                 h * 64 + kk + hi * 8);
      sc[t] = wmma_bf16(a, bb, sc[t]);
    }
  }

  // ---- masked softmax (row = query, reduce over 16 lanes of same half) ----
  const float scale = 0.125f;  // 1/sqrt(64)
#pragma unroll
  for (int r = 0; r < 8; ++r) {
    float v[5];
    float mx = -1e30f;
#pragma unroll
    for (int t = 0; t < 5; ++t) {
      int n = nl + 16 * t;
      v[t] = (n < 77) ? sc[t][r] * scale : -1e30f;
      mx = fmaxf(mx, v[t]);
    }
    for (int off = 1; off < 16; off <<= 1) mx = fmaxf(mx, __shfl_xor(mx, off, 32));
    float p[5];
    float sum = 0.f;
#pragma unroll
    for (int t = 0; t < 5; ++t) {
      int n = nl + 16 * t;
      p[t] = (n < 77) ? __expf(v[t] - mx) : 0.f;
      sum += p[t];
    }
    for (int off = 1; off < 16; off <<= 1) sum += __shfl_xor(sum, off, 32);
    float inv = 1.f / sum;
    int m = r + hi * 8;
#pragma unroll
    for (int t = 0; t < 5; ++t) P[w][m][nl + 16 * t] = (bf16_t)(p[t] * inv);
  }
  __syncthreads();

  // ---- out = P V ----
  v8f o[4] = {};
#pragma unroll
  for (int kt = 0; kt < 96; kt += 32) {
    v16bf a = load_a_frag(&P[w][nl][kt + hi * 8]);
#pragma unroll
    for (int j = 0; j < 4; ++j) {
      v16bf bb = load_a_frag(Vt + ((long)((b * 20 + h) * 64 + j * 16 + nl)) * 96 +
                                 kt + hi * 8);
      o[j] = wmma_bf16(a, bb, o[j]);
    }
  }

  // ---- epilogue: + v_ip (IP-adapter with 1 key => softmax == 1) ----
#pragma unroll
  for (int j = 0; j < 4; ++j)
#pragma unroll
    for (int r = 0; r < 8; ++r) {
      int m = r + hi * 8;
      int col = h * 64 + j * 16 + nl;
      long s = s0 + m;
      float val = o[j][r] + vip[b * 1280 + col];
      merged[((long)b * 4096 + s) * 1280 + col] = (bf16_t)val;
    }
}

// ---------------------------------------------------------------------------
// Small utility kernels
// ---------------------------------------------------------------------------
__global__ void cvt_f32_bf16(const float* __restrict__ in, bf16_t* __restrict__ out,
                             long n) {
  long i = (long)blockIdx.x * blockDim.x + threadIdx.x;
  long stride = (long)gridDim.x * blockDim.x;
  for (; i < n; i += stride) out[i] = (bf16_t)in[i];
}

// in: [K][N] f32 row-major -> out: [N][K] bf16 row-major
__global__ void wtrans_bf16(const float* __restrict__ in, bf16_t* __restrict__ out,
                            int K, int N) {
  long i = (long)blockIdx.x * blockDim.x + threadIdx.x;
  if (i >= (long)K * N) return;
  long n = i / K, k = i % K;
  out[i] = (bf16_t)in[k * (long)N + n];
}

// enc [B,77,2048] f32 -> [B,80,2048] bf16, rows 77..79 zero
__global__ void encpad_bf16(const float* __restrict__ enc, bf16_t* __restrict__ out) {
  long i = (long)blockIdx.x * blockDim.x + threadIdx.x;
  if (i >= 8L * 80 * 2048) return;
  int c = (int)(i % 2048);
  int r = (int)((i / 2048) % 80);
  int b = (int)(i / (2048L * 80));
  float v = (r < 77) ? enc[((long)(b * 77 + r)) * 2048 + c] : 0.f;
  out[i] = (bf16_t)v;
}

// V [B*80,1280] bf16 -> Vt [B,20,64,96] bf16 (k 80..95 zero; 77..79 already 0)
__global__ void vtrans_bf16(const bf16_t* __restrict__ V, bf16_t* __restrict__ Vt) {
  long i = (long)blockIdx.x * blockDim.x + threadIdx.x;
  if (i >= 8L * 20 * 64 * 96) return;
  int k = (int)(i % 96);
  int d = (int)((i / 96) % 64);
  int h = (int)((i / (96 * 64)) % 20);
  int b = (int)(i / (96L * 64 * 20));
  bf16_t val = (bf16_t)0.f;
  if (k < 80) val = V[((long)(b * 80 + k)) * 1280 + h * 64 + d];
  Vt[i] = val;
}

// v_ip[b,n] = adapter[b,:] @ Wv_adapter[:,n]  (f32, tiny)
__global__ void vip_gemv(const float* __restrict__ adapter,
                         const float* __restrict__ Wv_ad,
                         float* __restrict__ vip) {
  int i = blockIdx.x * blockDim.x + threadIdx.x;
  if (i >= 8 * 1280) return;
  int b = i / 1280, n = i % 1280;
  const float* a = adapter + b * 768;
  float s = 0.f;
  for (int k = 0; k < 768; ++k) s += a[k] * Wv_ad[(long)k * 1280 + n];
  vip[i] = s;
}

// ---------------------------------------------------------------------------
// Host launcher
// ---------------------------------------------------------------------------
extern "C" void kernel_launch(void* const* d_in, const int* in_sizes, int n_in,
                              void* d_out, int out_size, void* d_ws, size_t ws_size,
                              hipStream_t stream) {
  const float* hidden  = (const float*)d_in[0];  // [8,4096,1280]
  const float* enc     = (const float*)d_in[1];  // [8,77,2048]
  const float* adapter = (const float*)d_in[2];  // [8,1,768]
  const float* Wq      = (const float*)d_in[3];  // [1280,1280]
  const float* Wk      = (const float*)d_in[4];  // [2048,1280]
  const float* Wv      = (const float*)d_in[5];  // [2048,1280]
  // d_in[6] = Wk_adapter: unused (single-key softmax == 1)
  const float* Wv_ad   = (const float*)d_in[7];  // [768,1280]
  const float* Wo      = (const float*)d_in[8];  // [1280,1280]
  const float* bo      = (const float*)d_in[9];  // [1280]
  float* out = (float*)d_out;

  const long M = 8L * 4096;   // 32768
  const int  D = 1280;

  char* ws = (char*)d_ws;
  size_t off = 0;
  auto alloc = [&](size_t bytes) -> void* {
    void* p = ws + off;
    off = (off + bytes + 255) & ~(size_t)255;
    return p;
  };

  bf16_t* hid_bf = (bf16_t*)alloc(M * D * 2);          // later reused as merged
  bf16_t* q_bf   = (bf16_t*)alloc(M * D * 2);
  bf16_t* WqT    = (bf16_t*)alloc(1280L * 1280 * 2);
  bf16_t* WkT    = (bf16_t*)alloc(1280L * 2048 * 2);
  bf16_t* WvT    = (bf16_t*)alloc(1280L * 2048 * 2);
  bf16_t* WoT    = (bf16_t*)alloc(1280L * 1280 * 2);
  bf16_t* encp   = (bf16_t*)alloc(8L * 80 * 2048 * 2);
  bf16_t* Kb     = (bf16_t*)alloc(640L * 1280 * 2);
  bf16_t* Vb     = (bf16_t*)alloc(640L * 1280 * 2);
  bf16_t* Vt     = (bf16_t*)alloc(8L * 20 * 64 * 96 * 2);
  float*  vip    = (float*)alloc(8L * 1280 * 4);

  // 1) conversions / transposes
  cvt_f32_bf16<<<8192, 256, 0, stream>>>(hidden, hid_bf, M * D);
  wtrans_bf16<<<(1280L * 1280 + 255) / 256, 256, 0, stream>>>(Wq, WqT, 1280, 1280);
  wtrans_bf16<<<(2048L * 1280 + 255) / 256, 256, 0, stream>>>(Wk, WkT, 2048, 1280);
  wtrans_bf16<<<(2048L * 1280 + 255) / 256, 256, 0, stream>>>(Wv, WvT, 2048, 1280);
  wtrans_bf16<<<(1280L * 1280 + 255) / 256, 256, 0, stream>>>(Wo, WoT, 1280, 1280);
  encpad_bf16<<<(8L * 80 * 2048 + 255) / 256, 256, 0, stream>>>(enc, encp);

  // 2) projections (WMMA GEMMs, async-LDS staged B)
  gemm_bf16_wmma<true><<<dim3(D / 128, (int)(M / 128)), 256, 0, stream>>>(
      hid_bf, WqT, q_bf, nullptr, (int)M, D, 1280);
  gemm_bf16_wmma<true><<<dim3(D / 128, 640 / 128), 256, 0, stream>>>(
      encp, WkT, Kb, nullptr, 640, D, 2048);
  gemm_bf16_wmma<true><<<dim3(D / 128, 640 / 128), 256, 0, stream>>>(
      encp, WvT, Vb, nullptr, 640, D, 2048);

  // 3) per-head V transpose + IP-adapter value vector
  vtrans_bf16<<<(8L * 20 * 64 * 96 + 255) / 256, 256, 0, stream>>>(Vb, Vt);
  vip_gemv<<<(8 * 1280 + 255) / 256, 256, 0, stream>>>(adapter, Wv_ad, vip);

  // 4) fused attention + IP add -> merged (reuse hid_bf buffer)
  attn_wmma<<<dim3(4096 / 128, 20, 8), 256, 0, stream>>>(q_bf, Kb, Vt, vip, hid_bf);

  // 5) output projection with bias -> f32 out
  gemm_bf16_wmma<false><<<dim3(D / 128, (int)(M / 128)), 256, 0, stream>>>(
      hid_bf, WoT, out, bo, (int)M, D, 1280);
}
